// AxialTransformer_58600533786746
// MI455X (gfx1250) — matrix operand order repaired and checked by reading, not compiled, not run on hardware
//
#include <hip/hip_runtime.h>

// ---------------------------------------------------------------------------
// CDNA5 (gfx1250) implementation of the axial-transformer reference.
// All GEMMs use v_wmma_f32_16x16x32_bf16 (bf16 in, fp32 accumulate).
// LayerNorm / softmax / GELU / residuals in fp32. wave32 throughout.
// ---------------------------------------------------------------------------

typedef __attribute__((ext_vector_type(16))) __bf16 v16bf;
typedef __attribute__((ext_vector_type(8)))  float  v8f;

#define Tt 4
#define Ee 1024
#define Bb 8
#define Dd 128
#define Nn 128
#define Hh 8
#define FFdim 512
#define GTOK (Tt*Ee*Bb)      /* 32768 tokens in x      */
#define FTOK (Nn*Nn*Bb)      /* 131072 tokens in feats */
#define EBD  (Ee*Bb*Dd)      /* 1048576                */

__device__ __forceinline__ float bf2f(unsigned short u) {
  return __uint_as_float(((unsigned)u) << 16);
}
__device__ __forceinline__ unsigned short f2bf(float f) {
  unsigned u = __float_as_uint(f);
  unsigned r = u + 0x7fffu + ((u >> 16) & 1u);   // round to nearest even
  return (unsigned short)(r >> 16);
}
__device__ __forceinline__ __bf16 bfbits(unsigned short u) {
  union { unsigned short s; __bf16 b; } x; x.s = u; return x.b;
}
__device__ __forceinline__ v8f v8zero() {
  v8f z = {0.f,0.f,0.f,0.f,0.f,0.f,0.f,0.f}; return z;
}
__device__ __forceinline__ v8f wmma_bf16(v16bf a, v16bf b, v8f c) {
  return __builtin_amdgcn_wmma_f32_16x16x32_bf16(false, a, false, b, (short)0, c,
                                                 false, false);
}

// ---------------------------------------------------------------------------
// small utility kernels
// ---------------------------------------------------------------------------
__global__ void copy_f32_kernel(const float* __restrict__ s, float* __restrict__ d, long n) {
  long i = (long)blockIdx.x * 256 + threadIdx.x;
  if (i < n) d[i] = s[i];
}
__global__ void f2bf_kernel(const float* __restrict__ s, unsigned short* __restrict__ d, int n) {
  int i = blockIdx.x * 256 + threadIdx.x;
  if (i < n) d[i] = f2bf(s[i]);
}
__global__ void fill_i32_kernel(int* p, int n, int val) {
  int i = blockIdx.x * 256 + threadIdx.x;
  if (i < n) p[i] = val;
}

// ---------------------------------------------------------------------------
// LayerNorm over D=128, one row per wave (wave32, 4 elems/lane), bf16 output
// ---------------------------------------------------------------------------
__global__ void ln_kernel(const float* __restrict__ x, const float* __restrict__ g,
                          const float* __restrict__ b, unsigned short* __restrict__ out,
                          int rows) {
  int wid = threadIdx.x >> 5, lane = threadIdx.x & 31;
  long row = (long)blockIdx.x * 4 + wid;
  if (row >= rows) return;
  const float* xr = x + row * Dd;
  float v0 = xr[lane], v1 = xr[lane + 32], v2 = xr[lane + 64], v3 = xr[lane + 96];
  float s = v0 + v1 + v2 + v3;
#pragma unroll
  for (int off = 16; off; off >>= 1) s += __shfl_xor(s, off, 32);
  float mu = s * (1.f / 128.f);
  float d0 = v0 - mu, d1 = v1 - mu, d2 = v2 - mu, d3 = v3 - mu;
  float vs = d0*d0 + d1*d1 + d2*d2 + d3*d3;
#pragma unroll
  for (int off = 16; off; off >>= 1) vs += __shfl_xor(vs, off, 32);
  float inv = rsqrtf(vs * (1.f / 128.f) + 1e-12f);
  unsigned short* o = out + row * Dd;
  o[lane]      = f2bf(d0 * inv * g[lane]      + b[lane]);
  o[lane + 32] = f2bf(d1 * inv * g[lane + 32] + b[lane + 32]);
  o[lane + 64] = f2bf(d2 * inv * g[lane + 64] + b[lane + 64]);
  o[lane + 96] = f2bf(d3 * inv * g[lane + 96] + b[lane + 96]);
}

// ---------------------------------------------------------------------------
// Generic WMMA GEMM:  Y[t,n] = sum_k A[t,k]*W[n,k] (+bias) (+gelu) (+resid)
// A: bf16 [T,K] row-major, W: bf16 [Nout,K] row-major.
// One wave computes a 16x64 strip (4 accumulators sharing the A tile).
// ---------------------------------------------------------------------------
__global__ __launch_bounds__(256)
void gemm_kernel(const unsigned short* __restrict__ A, const unsigned short* __restrict__ W,
                 const float* __restrict__ bias, const float* __restrict__ resid,
                 float* __restrict__ outF, unsigned short* __restrict__ outBF,
                 int TT, int K, int NO, int act) {
  int wave = (blockIdx.x * blockDim.x + threadIdx.x) >> 5;
  int lane = threadIdx.x & 31;
  int lo = lane & 15, hi = lane >> 4;
  int strips_n = NO >> 6;
  int total = (TT >> 4) * strips_n;
  if (wave >= total) return;
  int tt = wave / strips_n, ns = wave % strips_n;

  v8f acc[4];
#pragma unroll
  for (int c = 0; c < 4; ++c) acc[c] = v8zero();

  int ksteps = K >> 5;
  for (int kk = 0; kk < ksteps; ++kk) {
    v16bf a;
    const unsigned short* arow = A + (size_t)(tt * 16 + lo) * K + kk * 32;
#pragma unroll
    for (int p = 0; p < 8; ++p) {
      // contiguous K layout: K pairs at laneHalf*16 + p*2
      int koff = (hi << 4) + (p << 1);
      unsigned u = *(const unsigned*)(arow + koff);
      a[2 * p]     = bfbits((unsigned short)u);
      a[2 * p + 1] = bfbits((unsigned short)(u >> 16));
    }
#pragma unroll
    for (int c = 0; c < 4; ++c) {
      int ncol = (ns << 6) + (c << 4) + lo;
      const unsigned short* brow = W + (size_t)ncol * K + kk * 32 + (hi << 4);
      v16bf b;
#pragma unroll
      for (int e = 0; e < 8; ++e) {
        unsigned u = *(const unsigned*)(brow + 2 * e);
        b[2 * e]     = bfbits((unsigned short)u);
        b[2 * e + 1] = bfbits((unsigned short)(u >> 16));
      }
      acc[c] = wmma_bf16(a, b, acc[c]);
    }
  }
#pragma unroll
  for (int c = 0; c < 4; ++c) {
    int ncol = (ns << 6) + (c << 4) + lo;
    float bv = bias ? bias[ncol] : 0.f;
#pragma unroll
    for (int vv = 0; vv < 8; ++vv) {
      int row = tt * 16 + (hi << 3) + vv;
      float val = acc[c][vv] + bv;
      if (act == 1) val = 0.5f * val * (1.f + erff(val * 0.70710678118654752f));
      size_t idx = (size_t)row * NO + ncol;
      if (resid) val += resid[idx];
      if (outF)  outF[idx]  = val;
      if (outBF) outBF[idx] = f2bf(val);
    }
  }
}

// ---------------------------------------------------------------------------
// Tied (MSA-style) axial attention, one wave per (head, batch, i-tile of 16).
//   S[i,j]   = sum_{r,d} Q[r,i,n,h,d] * K[r,j,n,h,d]        (K-dim = R*16)
//   P        = softmax_j(S)                                  (in LDS, bf16)
//   O[r,i,d] = sum_j P[i,j] * V[r,j,n,h,d]
// Logical (r,i) addressed through strides so row/col attention share code.
// ---------------------------------------------------------------------------
__global__ void tied_attn_kernel(const unsigned short* __restrict__ qb,
                                 const unsigned short* __restrict__ kb,
                                 const unsigned short* __restrict__ vb,
                                 unsigned short* __restrict__ cb,
                                 int R, int C, long sr, long si, long sn) {
  extern __shared__ unsigned short sm[];        // 16 rows x C cols, bf16
  const int lane = threadIdx.x & 31;
  const int lo = lane & 15, hi = lane >> 4;
  const int it = blockIdx.x;
  const int h = blockIdx.y & 7, n = blockIdx.y >> 3;
  const long base = (long)n * sn + h * 16;
  const int K = R * 16;

  // ---- phase 1: S tile (16 x C) via WMMA ----
  const long irow = it * 16 + lo;
  for (int jt = 0; jt < (C >> 4); ++jt) {
    v8f acc = v8zero();
    const long jcol = jt * 16 + lo;
    for (int kk = 0; kk < (K >> 5); ++kk) {
      v16bf a, b;
#pragma unroll
      for (int p = 0; p < 8; ++p) {
        int koff = (hi << 4) + (p << 1);
        int kidx = kk * 32 + koff;
        unsigned u = *(const unsigned*)(qb + (long)(kidx >> 4) * sr + irow * si + base + (kidx & 15));
        a[2 * p]     = bfbits((unsigned short)u);
        a[2 * p + 1] = bfbits((unsigned short)(u >> 16));
      }
      const unsigned short* kp = kb + (long)(kk * 2 + hi) * sr + jcol * si + base;
#pragma unroll
      for (int e = 0; e < 8; ++e) {
        unsigned u = *(const unsigned*)(kp + 2 * e);
        b[2 * e]     = bfbits((unsigned short)u);
        b[2 * e + 1] = bfbits((unsigned short)(u >> 16));
      }
      acc = wmma_bf16(a, b, acc);
    }
#pragma unroll
    for (int v = 0; v < 8; ++v)
      sm[(hi * 8 + v) * C + jt * 16 + lo] = f2bf(acc[v]);
  }
  __syncthreads();

  // ---- phase 2: softmax over j, in-wave (wave32 shuffles) ----
  for (int row = 0; row < 16; ++row) {
    float mx = -3.0e38f;
    for (int c = lane; c < C; c += 32) mx = fmaxf(mx, bf2f(sm[row * C + c]));
#pragma unroll
    for (int off = 16; off; off >>= 1) mx = fmaxf(mx, __shfl_xor(mx, off, 32));
    float sum = 0.f;
    for (int c = lane; c < C; c += 32) {
      unsigned short eb = f2bf(__expf(bf2f(sm[row * C + c]) - mx));
      sm[row * C + c] = eb;
      sum += bf2f(eb);
    }
#pragma unroll
    for (int off = 16; off; off >>= 1) sum += __shfl_xor(sum, off, 32);
    float inv = 1.f / sum;
    for (int c = lane; c < C; c += 32)
      sm[row * C + c] = f2bf(bf2f(sm[row * C + c]) * inv);
  }
  __syncthreads();

  // ---- phase 3: O = P * V, one r (16 d-cols) at a time ----
  for (int r = 0; r < R; ++r) {
    v8f o = v8zero();
    for (int jk = 0; jk < (C >> 5); ++jk) {
      v16bf a, b;
#pragma unroll
      for (int p = 0; p < 8; ++p) {
        int koff = (hi << 4) + (p << 1);
        int j0 = jk * 32 + koff;
        a[2 * p]     = bfbits(sm[lo * C + j0]);
        a[2 * p + 1] = bfbits(sm[lo * C + j0 + 1]);
      }
#pragma unroll
      for (int e = 0; e < 16; ++e) {
        int j = jk * 32 + hi * 16 + e;
        b[e] = bfbits(vb[(long)r * sr + (long)j * si + base + lo]);
      }
      o = wmma_bf16(a, b, o);
    }
#pragma unroll
    for (int vv = 0; vv < 8; ++vv) {
      long i = it * 16 + hi * 8 + vv;
      cb[(long)r * sr + i * si + base + lo] = f2bf(o[vv]);
    }
  }
}

// ---------------------------------------------------------------------------
// Graph column attention (seq len T=4, reduction E*hd=16384): reduction-bound,
// negligible FLOPs -> plain VALU kernels.
// probs layout: [h][n][i][j], j,i in [0,4)
// ---------------------------------------------------------------------------
__global__ void aw_small_kernel(const unsigned short* __restrict__ q,
                                const unsigned short* __restrict__ k,
                                float* __restrict__ probs) {
  int bid = blockIdx.x;                 // 256 blocks = h*32 + n*4 + i, decompose:
  int i = bid & 3, n = (bid >> 2) & 7, h = bid >> 5;
  int tid = threadIdx.x;
  int j = tid >> 6;                     // 4 groups of 64 threads
  int t = tid & 63;
  float s = 0.f;
  for (int f = t; f < Ee * 16; f += 64) {
    int e = f >> 4, d = f & 15;
    long off = ((long)e * Bb + n) * Dd + h * 16 + d;
    float qv = bf2f(q[(long)i * Ee * Bb * Dd + off]);
    float kv = bf2f(k[(long)j * Ee * Bb * Dd + off]);
    s += qv * kv;
  }
  __shared__ float red[256];
  red[tid] = s; __syncthreads();
  for (int st = 32; st; st >>= 1) { if (t < st) red[tid] += red[tid + st]; __syncthreads(); }
  if (tid == 0) {
    float a0 = red[0], a1 = red[64], a2 = red[128], a3 = red[192];
    float mx = fmaxf(fmaxf(a0, a1), fmaxf(a2, a3));
    float e0 = __expf(a0 - mx), e1 = __expf(a1 - mx), e2 = __expf(a2 - mx), e3 = __expf(a3 - mx);
    float inv = 1.f / (e0 + e1 + e2 + e3);
    float* p = probs + ((h * Bb + n) * 4 + i) * 4;
    p[0] = e0 * inv; p[1] = e1 * inv; p[2] = e2 * inv; p[3] = e3 * inv;
  }
}

__global__ void ctx_small_kernel(const float* __restrict__ probs,
                                 const unsigned short* __restrict__ v,
                                 unsigned short* __restrict__ ctx) {
  long f = (long)blockIdx.x * 256 + threadIdx.x;     // over T*E*B*D
  int dcol = f & 127;
  int n = (int)((f >> 7) & 7);
  long en = f >> 10;
  int e = (int)(en & 1023);
  int t = (int)(en >> 10);
  int h = dcol >> 4;
  const float* p = probs + ((h * Bb + n) * 4 + t) * 4;
  float s = 0.f;
#pragma unroll
  for (int j = 0; j < 4; ++j)
    s += p[j] * bf2f(v[(((long)j * Ee + e) * Bb + n) * Dd + dcol]);
  ctx[f] = f2bf(s);
}

// ---------------------------------------------------------------------------
// bridge kernels: mean over T, scatter (last-index-wins via atomicMax),
// concat builder, final gather + residual
// ---------------------------------------------------------------------------
__global__ void mean_kernel(const float* __restrict__ x, float* __restrict__ m) {
  long f = (long)blockIdx.x * 256 + threadIdx.x;     // over E*B*D
  float s = 0.f;
#pragma unroll
  for (int t = 0; t < Tt; ++t) s += x[(long)t * EBD + f];
  m[f] = s * 0.25f;
}
__global__ void scatter_kernel(const int* __restrict__ nodes, int* __restrict__ claim) {
  int e = blockIdx.x * 256 + threadIdx.x;
  if (e >= Ee) return;
  int ni = nodes[2 * e] - 1;     if (ni < 0) ni = 0;
  int nj = nodes[2 * e + 1] - 1; if (nj < 0) nj = 0;
  atomicMax(&claim[ni * Nn + nj], e);
}
__global__ void acat_kernel(const float* __restrict__ feats_in, const int* __restrict__ claim,
                            const float* __restrict__ m, const float* __restrict__ padv,
                            unsigned short* __restrict__ acat) {
  long f = (long)blockIdx.x * 256 + threadIdx.x;     // over FTOK*128
  int d = f & 127;
  long t = f >> 7;
  int n = (int)(t & 7);
  long cell = t >> 3;
  acat[t * 256 + d] = f2bf(feats_in[f]);
  int c = claim[cell];
  float val = (c >= 0) ? m[(long)c * (Bb * Dd) + n * Dd + d] : padv[0];
  acat[t * 256 + 128 + d] = f2bf(val);
}
__global__ void out_x_kernel(const float* __restrict__ x, const float* __restrict__ feats,
                             const int* __restrict__ nodes, float* __restrict__ out) {
  long f = (long)blockIdx.x * 256 + threadIdx.x;     // over T*E*B*D
  int d = f & 127;
  int n = (int)((f >> 7) & 7);
  int e = (int)((f >> 10) & 1023);
  int ni = nodes[2 * e] - 1;     if (ni < 0) ni = 0;
  int nj = nodes[2 * e + 1] - 1; if (nj < 0) nj = 0;
  out[f] = x[f] + feats[(((long)ni * Nn + nj) * Bb + n) * Dd + d];
}

// ---------------------------------------------------------------------------
// host orchestration
// ---------------------------------------------------------------------------
extern "C" void kernel_launch(void* const* d_in, const int* in_sizes, int n_in,
                              void* d_out, int out_size, void* d_ws, size_t ws_size,
                              hipStream_t stream) {
  (void)n_in; (void)out_size; (void)ws_size;

  // ---- resolve input pointers (supports both dict-insertion and jax-sorted) ----
  const bool sortedOrder = (in_sizes[0] != GTOK * Dd);   // feats-first => sorted keys
  auto P = [&](int i) { return (const float*)d_in[i]; };
  const float *in_x, *in_feats, *lin_w, *lin_b, *padv;
  const int* nodes;
  // field order used internally: [ln_g, ln_b, wq, bq, wk, bk, wv, bv, wo, bo]
  const float* A_[4][10];   // 0:graph.row 1:graph.col 2:feat.row 3:feat.col
  const float* F_[2][6];    // [ln_g, ln_b, w1, b1, w2, b2]; 0:graph.ffn 1:feat.ffn
  if (!sortedOrder) {
    in_x = P(0); in_feats = P(1); nodes = (const int*)d_in[2];
    int base[4] = {5, 15, 31, 41};
    for (int blk = 0; blk < 4; ++blk)
      for (int fld = 0; fld < 10; ++fld) A_[blk][fld] = P(base[blk] + fld);
    int fbase[2] = {25, 51};
    for (int blk = 0; blk < 2; ++blk)
      for (int fld = 0; fld < 6; ++fld) F_[blk][fld] = P(fbase[blk] + fld);
    padv = P(57); lin_w = P(58); lin_b = P(59);
  } else {
    in_feats = P(0); nodes = (const int*)d_in[1];
    int base[4] = {46, 30, 20, 4};             // graph.row, graph.col, feat.row, feat.col
    int perm[10] = {5, 4, 8, 2, 6, 0, 9, 3, 7, 1};   // sorted [bk,bo,bq,bv,ln_b,ln_g,wk,wo,wq,wv]
    for (int blk = 0; blk < 4; ++blk)
      for (int fld = 0; fld < 10; ++fld) A_[blk][fld] = P(base[blk] + perm[fld]);
    int fbase[2] = {40, 14};
    int fperm[6] = {3, 2, 4, 0, 5, 1};               // sorted [b1,b2,ln_b,ln_g,w1,w2]
    for (int blk = 0; blk < 2; ++blk)
      for (int fld = 0; fld < 6; ++fld) F_[blk][fld] = P(fbase[blk] + fperm[fld]);
    lin_b = P(56); lin_w = P(57); padv = P(58); in_x = P(59);
  }

  // ---- workspace arena ----
  char* wsb = (char*)d_ws;
  size_t off = 0;
  auto take = [&](size_t b) { size_t r = off; off += (b + 255) & ~(size_t)255; return r; };
  float*          x_cur     = (float*)(wsb + take((size_t)GTOK * Dd * 4));
  float*          feats_cur = (float*)(wsb + take((size_t)FTOK * Dd * 4));
  unsigned short* h_bf      = (unsigned short*)(wsb + take((size_t)FTOK * Dd * 2));
  unsigned short* q_bf      = (unsigned short*)(wsb + take((size_t)FTOK * Dd * 2)); // also Acat (q+k)
  unsigned short* k_bf      = (unsigned short*)(wsb + take((size_t)FTOK * Dd * 2));
  unsigned short* v_bf      = (unsigned short*)(wsb + take((size_t)FTOK * Dd * 2));
  unsigned short* ctx_bf    = (unsigned short*)(wsb + take((size_t)FTOK * Dd * 2));
  unsigned short* ffn_bf    = (unsigned short*)(wsb + take((size_t)16384 * FFdim * 2));
  float*          m_buf     = (float*)(wsb + take((size_t)EBD * 4));
  int*            claim     = (int*)(wsb + take((size_t)Nn * Nn * 4));
  float*          probs4    = (float*)(wsb + take((size_t)Hh * Bb * Tt * Tt * 4));
  unsigned short* wts       = (unsigned short*)(wsb + take((size_t)600000 * 2));
  (void)k_bf;

  // ---- weight conversion to bf16 arena ----
  unsigned short* wpos = wts;
  auto convW = [&](const float* src, int nelem) -> const unsigned short* {
    unsigned short* dst = wpos; wpos += nelem;
    f2bf_kernel<<<(nelem + 255) / 256, 256, 0, stream>>>(src, dst, nelem);
    return dst;
  };
  const unsigned short *Wq[4], *Wk[4], *Wv[4], *Wo[4];
  for (int i = 0; i < 4; ++i) {
    Wq[i] = convW(A_[i][2], Dd * Dd);
    Wk[i] = convW(A_[i][4], Dd * Dd);
    Wv[i] = convW(A_[i][6], Dd * Dd);
    Wo[i] = convW(A_[i][8], Dd * Dd);
  }
  const unsigned short *W1[2], *W2[2];
  for (int i = 0; i < 2; ++i) {
    W1[i] = convW(F_[i][2], FFdim * Dd);
    W2[i] = convW(F_[i][4], Dd * FFdim);
  }
  const unsigned short* Wlin = convW(lin_w, Dd * 2 * Dd);

  // ---- launch helpers ----
  auto run_gemm = [&](const unsigned short* A, const unsigned short* W, const float* bias,
                      const float* resid, float* outF, unsigned short* outBF,
                      int TT, int K, int NO, int act) {
    int total = (TT / 16) * (NO / 64);
    gemm_kernel<<<(total + 7) / 8, 256, 0, stream>>>(A, W, bias, resid, outF, outBF,
                                                     TT, K, NO, act);
  };
  auto attn_block = [&](int ab, float* xbuf, int tokens, bool isGraphCol,
                        int R, int C, long sr, long si) {
    ln_kernel<<<tokens / 4, 128, 0, stream>>>(xbuf, A_[ab][0], A_[ab][1], h_bf, tokens);
    run_gemm(h_bf, Wq[ab], A_[ab][3], nullptr, nullptr, q_bf, tokens, Dd, Dd, 0);
    run_gemm(h_bf, Wk[ab], A_[ab][5], nullptr, nullptr, k_bf, tokens, Dd, Dd, 0);
    run_gemm(h_bf, Wv[ab], A_[ab][7], nullptr, nullptr, v_bf, tokens, Dd, Dd, 0);
    if (isGraphCol) {
      aw_small_kernel<<<256, 256, 0, stream>>>(q_bf, k_bf, probs4);
      ctx_small_kernel<<<(GTOK * Dd) / 256, 256, 0, stream>>>(probs4, v_bf, ctx_bf);
    } else {
      tied_attn_kernel<<<dim3(C / 16, Hh * Bb), 32, (size_t)16 * C * 2, stream>>>(
          q_bf, k_bf, v_bf, ctx_bf, R, C, sr, si, (long)Dd);
    }
    run_gemm(ctx_bf, Wo[ab], A_[ab][9], xbuf, xbuf, nullptr, tokens, Dd, Dd, 0);
  };
  auto ffn_block = [&](int fb, float* xbuf, int tokens) {
    ln_kernel<<<tokens / 4, 128, 0, stream>>>(xbuf, F_[fb][0], F_[fb][1], h_bf, tokens);
    const int CH = 16384;
    for (int c0 = 0; c0 < tokens; c0 += CH) {
      run_gemm(h_bf + (size_t)c0 * Dd, W1[fb], F_[fb][3], nullptr, nullptr, ffn_bf,
               CH, Dd, FFdim, 1);
      run_gemm(ffn_bf, W2[fb], F_[fb][5], xbuf + (size_t)c0 * Dd, xbuf + (size_t)c0 * Dd,
               nullptr, CH, FFdim, Dd, 0);
    }
  };

  // ---- graph axial layer on x [T,E,B,D] ----
  copy_f32_kernel<<<(GTOK * Dd) / 256, 256, 0, stream>>>(in_x, x_cur, (long)GTOK * Dd);
  // row attention: attend over E, tied over T (K = T*16 = 64)
  attn_block(0, x_cur, GTOK, false, Tt, Ee, (long)Ee * Bb * Dd, (long)Bb * Dd);
  // col attention: attend over T (len 4), tied over E -> VALU path
  attn_block(1, x_cur, GTOK, true, 0, 0, 0, 0);
  ffn_block(0, x_cur, GTOK);

  // ---- bridge: mean over T, scatter into padded grid, concat, lin projection ----
  mean_kernel<<<EBD / 256, 256, 0, stream>>>(x_cur, m_buf);
  fill_i32_kernel<<<(Nn * Nn) / 256, 256, 0, stream>>>(claim, Nn * Nn, -1);
  scatter_kernel<<<4, 256, 0, stream>>>(nodes, claim);
  acat_kernel<<<(FTOK * Dd) / 256, 256, 0, stream>>>(in_feats, claim, m_buf, padv, q_bf);
  run_gemm(q_bf /*Acat spans q+k slots*/, Wlin, lin_b, nullptr, feats_cur, nullptr,
           FTOK, 2 * Dd, Dd, 0);

  // ---- feat axial layer on feats [N,N,B,D] ----
  attn_block(2, feats_cur, FTOK, false, Nn, Nn, (long)Nn * Bb * Dd, (long)Bb * Dd);
  attn_block(3, feats_cur, FTOK, false, Nn, Nn, (long)Bb * Dd, (long)Nn * Bb * Dd);
  ffn_block(1, feats_cur, FTOK);

  // ---- outputs: (x + gathered feats, feats) ----
  out_x_kernel<<<(GTOK * Dd) / 256, 256, 0, stream>>>(x_cur, feats_cur, nodes, (float*)d_out);
  copy_f32_kernel<<<(FTOK * Dd) / 256, 256, 0, stream>>>(feats_cur,
      (float*)d_out + (size_t)GTOK * Dd, (long)FTOK * Dd);
}
